// SelfAttention_34514357191295
// MI455X (gfx1250) — matrix operand order repaired and checked by reading, not compiled, not run on hardware
//
#include <hip/hip_runtime.h>
#include <hip/hip_bf16.h>

typedef __attribute__((ext_vector_type(16))) _Float16 v16h;
typedef __attribute__((ext_vector_type(8)))  _Float16 v8h;
typedef __attribute__((ext_vector_type(4)))  _Float16 v4h;
typedef __attribute__((ext_vector_type(8)))  float    v8f;
typedef __attribute__((ext_vector_type(4)))  int      v4i;

#define SEQ    4096
#define DMODEL 768
#define D3     2304
#define NHEAD  12
#define HD     64

#if __has_builtin(__builtin_amdgcn_global_load_async_to_lds_b128) && \
    __has_builtin(__builtin_amdgcn_s_wait_asynccnt)
#define USE_ASYNC_LDS 1
#endif

#ifdef USE_ASYNC_LDS
static __device__ __forceinline__ void async_g2l_b128(const void* g, void* l) {
  __builtin_amdgcn_global_load_async_to_lds_b128(
      (__attribute__((address_space(1))) v4i*)g,
      (__attribute__((address_space(3))) v4i*)l, 0, 0);
}
#endif

static __device__ __forceinline__ v8f wmma_f16(v16h a, v16h b, v8f c) {
  // D = A(16x32 f16) * B(32x16 f16) + C(16x16 f32)
  return __builtin_amdgcn_wmma_f32_16x16x32_f16(false, a, false, b, (short)0, c,
                                                false, false);
}

static __device__ __forceinline__ v16h pack16(v8h lo8, v8h hi8) {
  v16h r;
#pragma unroll
  for (int e = 0; e < 8; ++e) { r[e] = lo8[e]; r[e + 8] = hi8[e]; }
  return r;
}

// max/sum over the 16-lane half-group (xor masks < 16 stay in-group)
static __device__ __forceinline__ float redmax16(float v) {
#pragma unroll
  for (int m = 8; m >= 1; m >>= 1) v = fmaxf(v, __shfl_xor(v, m, 32));
  return v;
}
static __device__ __forceinline__ float redsum16(float v) {
#pragma unroll
  for (int m = 8; m >= 1; m >>= 1) v += __shfl_xor(v, m, 32);
  return v;
}

// ---------------------------------------------------------------------------
// C[M,N] = X[M,K] @ W[N,K]^T + bias[N]; f32 inputs, f16 or f32 output.
// Block tile 128x128, K-step 32, 8 waves (4 M x 2 N), each wave 32x64.
// A fragment (16x32 f16, lane l, elem e): M=l&15, K=8*(l>>4)+(e&7)+16*(e>>3)
// B fragment (32x16 f16): N=l&15, K=8*(l>>4)+(e&7)+16*(e>>3)
// C/D (16x16 f32): vgpr i -> M = i + 8*(l>>4), N = l&15
// ---------------------------------------------------------------------------
__global__ void __launch_bounds__(256)
gemm_xwT_bias(const float* __restrict__ X, const float* __restrict__ W,
              const float* __restrict__ bias, int M, int N, int K,
              float* __restrict__ outF, _Float16* __restrict__ outH) {
  __shared__ alignas(16) _Float16 lx[128][32];  // 8 KB  [m][k]
  __shared__ alignas(16) _Float16 lw[128][32];  // 8 KB  [n][k]

  const int tid  = threadIdx.x;
  const int lane = tid & 31;
  const int wave = tid >> 5;        // 0..7
  const int wm   = wave & 3;        // M sub-tile: wm*32
  const int wn   = wave >> 2;       // N sub-tile: wn*64
  const int bm0  = blockIdx.x * 128;
  const int bn0  = blockIdx.y * 128;
  const int hi   = lane >> 4;
  const int lo   = lane & 15;

  v8f c[2][4] = {};

  for (int k0 = 0; k0 < K; k0 += 32) {
    // Issue all tile loads (b128) before the barrier: overlaps previous math.
    float4 xv[4], wv[4];
#pragma unroll
    for (int p = 0; p < 4; ++p) {
      int idx = p * 1024 + tid * 4;          // 128*32 elems, 4 per thread/pass
      int r = idx >> 5, kk = idx & 31;
      xv[p] = *(const float4*)&X[(size_t)(bm0 + r) * K + k0 + kk];
      wv[p] = *(const float4*)&W[(size_t)(bn0 + r) * K + k0 + kk];
    }
    __syncthreads();                          // prior iteration done reading LDS
#pragma unroll
    for (int p = 0; p < 4; ++p) {
      int idx = p * 1024 + tid * 4;
      int r = idx >> 5, kk = idx & 31;
      v4h xh = {(_Float16)xv[p].x, (_Float16)xv[p].y,
                (_Float16)xv[p].z, (_Float16)xv[p].w};
      v4h wh = {(_Float16)wv[p].x, (_Float16)wv[p].y,
                (_Float16)wv[p].z, (_Float16)wv[p].w};
      *(v4h*)&lx[r][kk] = xh;
      *(v4h*)&lw[r][kk] = wh;
    }
    __syncthreads();

    v16h a[2];
#pragma unroll
    for (int mi = 0; mi < 2; ++mi) {
      int m  = wm * 32 + mi * 16 + lo;
      int kb = hi * 8;
      v8h l8 = *(const v8h*)&lx[m][kb];
      v8h h8 = *(const v8h*)&lx[m][kb + 16];
      a[mi] = pack16(l8, h8);
    }
#pragma unroll
    for (int nj = 0; nj < 4; ++nj) {
      int n  = wn * 64 + nj * 16 + lo;
      int kb = hi * 8;
      v8h b0 = *(const v8h*)&lw[n][kb];
      v8h b1 = *(const v8h*)&lw[n][kb + 16];
      v16h b = pack16(b0, b1);
#pragma unroll
      for (int mi = 0; mi < 2; ++mi) c[mi][nj] = wmma_f16(a[mi], b, c[mi][nj]);
    }
  }

#pragma unroll
  for (int mi = 0; mi < 2; ++mi)
#pragma unroll
    for (int nj = 0; nj < 4; ++nj) {
      int ncol = bn0 + wn * 64 + nj * 16 + lo;
      float bv = bias[ncol];
#pragma unroll
      for (int i = 0; i < 8; ++i) {
        int row = bm0 + wm * 32 + mi * 16 + hi * 8 + i;
        float v = c[mi][nj][i] + bv;
        if (outH) outH[(size_t)row * N + ncol] = (_Float16)v;
        else      outF[(size_t)row * N + ncol] = v;
      }
    }
}

// ---------------------------------------------------------------------------
// Flash attention. qkv: f16 [S][3*D] (Q at col h*64, K at 768+h*64, V at
// 1536+h*64). Each wave: one 16-row query tile; 8 waves/block share one head
// so K/V 32x64 tiles are staged once in LDS per 32-key step. K tile staged
// via async global->LDS (ASYNCcnt) when available.
// ---------------------------------------------------------------------------
__global__ void __launch_bounds__(256)
attn_flash(const _Float16* __restrict__ qkv, float* __restrict__ attn) {
  __shared__ alignas(16) _Float16 lk[32][64];        // 4 KB  [key][feat]
  __shared__ alignas(16) _Float16 lv[64][32];        // 4 KB  [feat][key] (V^T)
  __shared__ alignas(16) _Float16 pbuf[8][16][32];   // 8 KB  per-wave P tile

  const int tid  = threadIdx.x;
  const int lane = tid & 31;
  const int wave = tid >> 5;
  const int head = blockIdx.x >> 5;        // 12 heads x 32 q-groups
  const int qg   = blockIdx.x & 31;
  const int q0   = (qg * 8 + wave) * 16;   // first query row of this wave
  const int hb   = head * HD;
  const int hi   = lane >> 4;
  const int lo   = lane & 15;
  const int skk  = tid >> 3;               // staging: key   0..31
  const int sff  = (tid & 7) * 8;          // staging: feat  0,8,..,56

  // Q fragments: 16x64 split into two 16x32 A fragments (feature chunks)
  v16h aq[2];
  {
    const _Float16* qrow = qkv + (size_t)(q0 + lo) * D3 + hb;
#pragma unroll
    for (int cch = 0; cch < 2; ++cch) {
      int kb = cch * 32 + hi * 8;
      v8h l8 = *(const v8h*)(qrow + kb);
      v8h h8 = *(const v8h*)(qrow + kb + 16);
      aq[cch] = pack16(l8, h8);
    }
  }

  v8f o[4] = {};
  float mrow[8], lrow[8];
#pragma unroll
  for (int i = 0; i < 8; ++i) { mrow[i] = -3.0e38f; lrow[i] = 0.0f; }

  for (int kb0 = 0; kb0 < SEQ; kb0 += 32) {
    const size_t rowb = (size_t)(kb0 + skk) * D3 + hb + sff;
    // V (and K on the fallback path) loaded to registers before the barrier.
    v8h vv = *(const v8h*)(qkv + rowb + 2 * DMODEL);
#ifndef USE_ASYNC_LDS
    v8h kv8 = *(const v8h*)(qkv + rowb + DMODEL);
#endif
    __syncthreads();                         // prior iteration done with lk/lv
#ifdef USE_ASYNC_LDS
    async_g2l_b128(qkv + rowb + DMODEL, &lk[skk][sff]);   // 16B per lane
#else
    *(v8h*)&lk[skk][sff] = kv8;
#endif
#pragma unroll
    for (int e = 0; e < 8; ++e) lv[sff + e][skk] = vv[e]; // V transpose
#ifdef USE_ASYNC_LDS
    __builtin_amdgcn_s_wait_asynccnt(0);
#endif
    __syncthreads();

    // scores: two key halves x two feature chunks
    v8f s[2] = {};
#pragma unroll
    for (int kh = 0; kh < 2; ++kh) {
      int n = kh * 16 + lo;                  // key within block
#pragma unroll
      for (int cch = 0; cch < 2; ++cch) {
        int fb = cch * 32 + hi * 8;
        v8h b0 = *(const v8h*)&lk[n][fb];
        v8h b1 = *(const v8h*)&lk[n][fb + 16];
        s[kh] = wmma_f16(aq[cch], pack16(b0, b1), s[kh]);
      }
    }

    // online softmax per row (row = hi*8 + i, columns spread over 16 lanes)
#pragma unroll
    for (int i = 0; i < 8; ++i) {
      float s0 = s[0][i] * 0.125f;           // 1/sqrt(64)
      float s1 = s[1][i] * 0.125f;
      float mx   = redmax16(fmaxf(s0, s1));
      float newm = fmaxf(mrow[i], mx);
      float alpha = __expf(mrow[i] - newm);
      float p0 = __expf(s0 - newm);
      float p1 = __expf(s1 - newm);
      float rs = redsum16(p0 + p1);
      lrow[i] = lrow[i] * alpha + rs;
      mrow[i] = newm;
#pragma unroll
      for (int j = 0; j < 4; ++j) o[j][i] *= alpha;
      pbuf[wave][hi * 8 + i][lo]      = (_Float16)p0;
      pbuf[wave][hi * 8 + i][16 + lo] = (_Float16)p1;
    }
    // pbuf is per-wave and LDS ops from one wave execute in order; only stop
    // the compiler from reordering the transpose read before the writes.
    asm volatile("" ::: "memory");

    // P as A fragment (16 rows x 32 keys)
    int kb2 = hi * 8;
    v8h pa0 = *(const v8h*)&pbuf[wave][lo][kb2];
    v8h pa1 = *(const v8h*)&pbuf[wave][lo][kb2 + 16];
    v16h pa = pack16(pa0, pa1);

    // O += P @ V  (four 16-wide feature groups)
#pragma unroll
    for (int j = 0; j < 4; ++j) {
      int n  = j * 16 + lo;                   // output feature
      int kk = hi * 8;                        // key within block
      v8h v0 = *(const v8h*)&lv[n][kk];
      v8h v1 = *(const v8h*)&lv[n][kk + 16];
      o[j] = wmma_f16(pa, pack16(v0, v1), o[j]);
    }
  }

#pragma unroll
  for (int j = 0; j < 4; ++j)
#pragma unroll
    for (int i = 0; i < 8; ++i) {
      int row = q0 + hi * 8 + i;
      int col = hb + j * 16 + lo;
      attn[(size_t)row * DMODEL + col] = o[j][i] / lrow[i];
    }
}

extern "C" void kernel_launch(void* const* d_in, const int* in_sizes, int n_in,
                              void* d_out, int out_size, void* d_ws, size_t ws_size,
                              hipStream_t stream) {
  (void)in_sizes; (void)n_in; (void)out_size; (void)ws_size;
  const float* x     = (const float*)d_in[0];
  const float* w_qkv = (const float*)d_in[1];
  const float* b_qkv = (const float*)d_in[2];
  const float* w_out = (const float*)d_in[3];
  const float* b_out = (const float*)d_in[4];
  float* out = (float*)d_out;

  _Float16* qkv  = (_Float16*)d_ws;                                     // 18.9 MB
  float*    attn = (float*)((char*)d_ws + (size_t)SEQ * D3 * sizeof(_Float16));

  // 1) QKV projection: [4096,2304] = x @ w_qkv^T + b_qkv, stored f16
  dim3 g1(SEQ / 128, D3 / 128);
  gemm_xwT_bias<<<g1, 256, 0, stream>>>(x, w_qkv, b_qkv, SEQ, D3, DMODEL,
                                        nullptr, qkv);
  // 2) Flash attention -> attn [4096,768] f32
  attn_flash<<<NHEAD * 32, 256, 0, stream>>>(qkv, attn);
  // 3) Output projection: out = attn @ w_out^T + b_out
  dim3 g3(SEQ / 128, DMODEL / 128);
  gemm_xwT_bias<<<g3, 256, 0, stream>>>(attn, w_out, b_out, SEQ, DMODEL, DMODEL,
                                        out, nullptr);
}
